// RGCNWithLearnableAdj_43550968382025
// MI455X (gfx1250) — matrix operand order repaired and checked by reading, not compiled, not run on hardware
//
#include <hip/hip_runtime.h>
#include <hip/hip_bf16.h>
#include <stdint.h>

// Problem constants (fixed by the reference)
#define N_NODES 65536
#define KDIM    512      // contraction dim of BOTH gemms (IN_DIM == HID_DIM == 512)
#define HID     512
#define ODIM    256
#define CHUNKS  64       // row chunks for deterministic column-sum reduction

typedef __bf16 v16b __attribute__((ext_vector_type(16)));
typedef __bf16 v8b  __attribute__((ext_vector_type(8)));
typedef __bf16 v2b  __attribute__((ext_vector_type(2)));
typedef float  v8f  __attribute__((ext_vector_type(8)));

#if defined(__has_builtin) && __has_builtin(__builtin_amdgcn_sched_barrier)
#define SCHED_FENCE() __builtin_amdgcn_sched_barrier(0)
#else
#define SCHED_FENCE()
#endif

// concatenate two 8-lane halves into a 16-lane fragment (register-level only)
__device__ __forceinline__ v16b cat8(v8b lo, v8b hi) {
  return __builtin_shufflevector(lo, hi, 0, 1, 2, 3, 4, 5, 6, 7,
                                 8, 9, 10, 11, 12, 13, 14, 15);
}

__device__ __forceinline__ float fget(const float4& v, int j) {
  union { float4 v; float f[4]; } u; u.v = v; return u.f[j];
}

// ---------------------------------------------------------------------------
// Pass 1a: partial column sums (deterministic, no float atomics).
// grid = (cols/256, CHUNKS), block = 256. X is [N_NODES][512].
// ---------------------------------------------------------------------------
__global__ __launch_bounds__(256)
void colsum_partial(const float* __restrict__ X, float* __restrict__ part,
                    int rows_per_chunk) {
  const int col = blockIdx.x * blockDim.x + threadIdx.x;     // < 512
  const int r0  = blockIdx.y * rows_per_chunk;
  float s = 0.f;
  for (int r = 0; r < rows_per_chunk; ++r)
    s += X[(size_t)(r0 + r) * KDIM + col];                   // coalesced per row
  part[(size_t)blockIdx.y * KDIM + col] = s;
}

// Pass 1b: reduce CHUNKS partials per column (fixed order -> deterministic).
__global__ __launch_bounds__(256)
void colsum_reduce(const float* __restrict__ part, float* __restrict__ s) {
  const int col = blockIdx.x * blockDim.x + threadIdx.x;
  float acc = 0.f;
  for (int c = 0; c < CHUNKS; ++c) acc += part[(size_t)c * KDIM + col];
  s[col] = acc;
}

// ---------------------------------------------------------------------------
// Relational aggregate (tiny):  agg[o] = bias[o] + sum_b (sum_r comp[r,b]) *
//                                         (sum_i s[i] * V[b,i,o])
// V is [NB][512][OUT] row-major, comp is [R][NB].
// ---------------------------------------------------------------------------
__global__ __launch_bounds__(256)
void agg_kernel(const float* __restrict__ s, const float* __restrict__ V,
                const float* __restrict__ comp, const float* __restrict__ bias,
                float* __restrict__ agg, int NB, int R, int OUT) {
  const int o = blockIdx.x * blockDim.x + threadIdx.x;
  if (o >= OUT) return;
  float acc = bias[o];
  for (int b = 0; b < NB; ++b) {
    float cb = 0.f;
    for (int r = 0; r < R; ++r) cb += comp[r * NB + b];
    float sv = 0.f;
    for (int i = 0; i < KDIM; ++i)
      sv += s[i] * V[((size_t)b * KDIM + i) * OUT + o];       // coalesced in o
    acc += cb * sv;
  }
  agg[o] = acc;
}

// ---------------------------------------------------------------------------
// Main GEMM:  C[m,n] = act( A[m,:] @ W[:,n] + agg[n] )
//   A: [N_NODES][512] f32, W: [512][Nout] f32; f32 accumulate via
//   V_WMMA_F32_16X16X32_BF16 (A/W converted to bf16 while staging into LDS).
// Block tile 128(M) x 64(N), 8 waves, each wave owns a 32x32 sub-tile
// (2x2 of 16x16 WMMA accumulators). K staged 64 wide (2 sub-steps of 32 ->
// 8 WMMAs per barrier), double-buffered LDS, software-pipelined:
//     gload(next stage, regs) | sched_barrier | compute(cur) -> lstore(next)
// The sched_barrier(0) pins all 12 global_load_b128 BEFORE the WMMAs so
// their latency hides behind the matrix pipe. B staging assigns each thread
// two adjacent K-rows so every transposed (k,k+1) bf16 pair is ONE aligned
// ds_store_b32 fed directly by v_cvt_pk_bf16_f32 (no repacking moves).
// LDS layouts: As[M][K], Bs[N][K] so EVERY fragment load is a contiguous
// 16B ds_load_b128. Row stride 72 bf16 = 144 B keeps 16B alignment and
// spreads banks. 54 KB/block, 2 blocks/WGP of 320 KB.
// ---------------------------------------------------------------------------
template <bool RELU>
__global__ __launch_bounds__(256, 2)
void rgcn_gemm_bf16(const float* __restrict__ A, const float* __restrict__ W,
                    const float* __restrict__ agg, float* __restrict__ C,
                    int Nout) {
  constexpr int KB  = 64;        // K per stage
  constexpr int LDK = KB + 8;    // 72 elems = 144 B rows (16B-aligned, even)
  __shared__ __attribute__((aligned(16))) __bf16 As[2][128][LDK];
  __shared__ __attribute__((aligned(16))) __bf16 Bs[2][64][LDK];

  const int tid  = threadIdx.x;
  const int lane = tid & 31;            // gfx1250 is wave32-only
  const int wave = tid >> 5;            // 8 waves / block
  const int wm   = (wave & 3) * 32;     // wave M offset in block tile
  const int wn   = (wave >> 2) * 32;    // wave N offset in block tile

  const int m0 = blockIdx.y * 128;
  const int n0 = blockIdx.x * 64;

  // staging thread mapping: 16 float4 per 64-wide row
  const int srow = tid >> 4;            // 0..15
  const int sc4  = (tid & 15) << 2;     // 0,4,...,60

  v8f c00 = {}, c01 = {}, c10 = {}, c11 = {};
  float4 ra[8];                         // staged A values (held across WMMAs)
  float4 rb[4];                         // staged W values: K-row pairs

  // issue all global loads of one stage; results consumed later by lstore()
  auto gload = [&](int k0) {
#pragma unroll
    for (int i = 0; i < 8; ++i)         // A tile: 128 x 64 f32
      ra[i] = *reinterpret_cast<const float4*>(
          &A[(size_t)(m0 + srow + i * 16) * KDIM + k0 + sc4]);
#pragma unroll
    for (int i = 0; i < 2; ++i) {       // W tile: 64 x 64 f32, rows 2s, 2s+1
      const int krow = i * 32 + 2 * srow;
      rb[2 * i + 0] = *reinterpret_cast<const float4*>(
          &W[(size_t)(k0 + krow) * Nout + n0 + sc4]);
      rb[2 * i + 1] = *reinterpret_cast<const float4*>(
          &W[(size_t)(k0 + krow + 1) * Nout + n0 + sc4]);
    }
  };

  // convert staged registers to bf16 and store into LDS buffer `buf`
  auto lstore = [&](int buf) {
#pragma unroll
    for (int i = 0; i < 8; ++i) {       // A: 4 contiguous bf16 -> b64 store
      const int row = srow + i * 16;
      As[buf][row][sc4 + 0] = (__bf16)ra[i].x;
      As[buf][row][sc4 + 1] = (__bf16)ra[i].y;
      As[buf][row][sc4 + 2] = (__bf16)ra[i].z;
      As[buf][row][sc4 + 3] = (__bf16)ra[i].w;
    }
#pragma unroll
    for (int i = 0; i < 2; ++i) {       // B transpose: (k,k+1) pair = b32 store
      const int krow = i * 32 + 2 * srow;
#pragma unroll
      for (int j = 0; j < 4; ++j) {
        v2b p;
        p[0] = (__bf16)fget(rb[2 * i + 0], j);   // K = krow
        p[1] = (__bf16)fget(rb[2 * i + 1], j);   // K = krow+1 (contiguous)
        *reinterpret_cast<v2b*>(&Bs[buf][sc4 + j][krow]) = p;
      }
    }
  };

  // fragment index helpers (ISA 7.12.2, 16-bit A layout; B mirrors with N=lane)
  const int mrow = lane & 15;
  const int kb   = (lane >> 4) * 8;     // A: K-base 0 / 8 by lane half
  const int kh   = (lane >> 4) * 8;     // B: K-base 0 / 8 by lane half (same as A)

  auto compute = [&](int buf) {
#pragma unroll
    for (int kk = 0; kk < KB; kk += 32) {
      const __bf16* pa0 = &As[buf][wm + mrow][kk];
      const __bf16* pa1 = &As[buf][wm + 16 + mrow][kk];
      const v16b a0 = cat8(*(const v8b*)(pa0 + kb), *(const v8b*)(pa0 + kb + 16));
      const v16b a1 = cat8(*(const v8b*)(pa1 + kb), *(const v8b*)(pa1 + kb + 16));

      const __bf16* pb0 = &Bs[buf][wn + mrow][kk + kh];
      const __bf16* pb1 = &Bs[buf][wn + 16 + mrow][kk + kh];
      const v16b b0 = cat8(*(const v8b*)pb0, *(const v8b*)(pb0 + 16));
      const v16b b1 = cat8(*(const v8b*)pb1, *(const v8b*)(pb1 + 16));

      c00 = __builtin_amdgcn_wmma_f32_16x16x32_bf16(false, a0, false, b0,
                                                    (short)0, c00, false, false);
      c01 = __builtin_amdgcn_wmma_f32_16x16x32_bf16(false, a0, false, b1,
                                                    (short)0, c01, false, false);
      c10 = __builtin_amdgcn_wmma_f32_16x16x32_bf16(false, a1, false, b0,
                                                    (short)0, c10, false, false);
      c11 = __builtin_amdgcn_wmma_f32_16x16x32_bf16(false, a1, false, b1,
                                                    (short)0, c11, false, false);
    }
  };

  // ---- software pipeline over KDIM/KB = 8 stages ----
  gload(0);
  lstore(0);
  __syncthreads();
  int buf = 0;
  for (int s = 1; s < KDIM / KB; ++s) {
    gload(s * KB);          // 12 loads issued up-front...
    SCHED_FENCE();          // ...and pinned before the WMMAs below
    compute(buf);           // 8 WMMAs hide the global-load latency
    lstore(buf ^ 1);        // waits + converts + LDS stores
    __syncthreads();
    buf ^= 1;
  }
  compute(buf);             // last stage, no successor

  // ---- epilogue: D layout lane L, VGPR v -> (M = 8*(L>=16)+v, N = L&15) ----
  const int mhalf = (lane >> 4) * 8;
  const int ncol  = lane & 15;
#pragma unroll
  for (int mt = 0; mt < 2; ++mt) {
#pragma unroll
    for (int nt = 0; nt < 2; ++nt) {
      const v8f acc = (mt == 0) ? (nt == 0 ? c00 : c01)
                                : (nt == 0 ? c10 : c11);
      const int gm = m0 + wm + mt * 16 + mhalf;
      const int gn = n0 + wn + nt * 16 + ncol;
      const float add = agg[gn];
#pragma unroll
      for (int v = 0; v < 8; ++v) {
        float r = acc[v] + add;
        if (RELU) r = r > 0.f ? r : 0.f;
        C[(size_t)(gm + v) * Nout + gn] = r;
      }
    }
  }
}

// ---------------------------------------------------------------------------
// Launch: 8 stream-ordered kernels.
// Workspace layout (needs ~128.3 MB):
//   h      : 65536*512 f32 (layer-1 activations)
//   aux    : s1[512] agg1[512] s2[512] agg2[256(+pad)] partials[CHUNKS*512]
// ---------------------------------------------------------------------------
extern "C" void kernel_launch(void* const* d_in, const int* in_sizes, int n_in,
                              void* d_out, int out_size, void* d_ws, size_t ws_size,
                              hipStream_t stream) {
  (void)in_sizes; (void)n_in; (void)out_size; (void)ws_size;

  const float* x     = (const float*)d_in[0];
  const float* V1    = (const float*)d_in[1];
  const float* comp1 = (const float*)d_in[2];
  const float* loop1 = (const float*)d_in[3];
  const float* bias1 = (const float*)d_in[4];
  const float* V2    = (const float*)d_in[5];
  const float* comp2 = (const float*)d_in[6];
  const float* loop2 = (const float*)d_in[7];
  const float* bias2 = (const float*)d_in[8];
  float* out = (float*)d_out;

  float* h    = (float*)d_ws;                       // 65536*512
  float* aux  = h + (size_t)N_NODES * HID;
  float* s1   = aux;                                // 512
  float* agg1 = aux + 512;                          // 512
  float* s2   = aux + 1024;                         // 512
  float* agg2 = aux + 1536;                         // 256 (+pad to 512)
  float* part = aux + 2048;                         // CHUNKS*512

  const int rows_per_chunk = N_NODES / CHUNKS;      // 1024
  const dim3 blk(256);

  // ---- layer 1 ----
  colsum_partial<<<dim3(KDIM / 256, CHUNKS), blk, 0, stream>>>(x, part, rows_per_chunk);
  colsum_reduce <<<dim3(KDIM / 256), blk, 0, stream>>>(part, s1);
  agg_kernel    <<<dim3(HID / 256),  blk, 0, stream>>>(s1, V1, comp1, bias1, agg1,
                                                       /*NB=*/4, /*R=*/16, /*OUT=*/HID);
  rgcn_gemm_bf16<true><<<dim3(HID / 64, N_NODES / 128), blk, 0, stream>>>(
      x, loop1, agg1, h, HID);

  // ---- layer 2 ----
  colsum_partial<<<dim3(KDIM / 256, CHUNKS), blk, 0, stream>>>(h, part, rows_per_chunk);
  colsum_reduce <<<dim3(KDIM / 256), blk, 0, stream>>>(part, s2);
  agg_kernel    <<<dim3(1), blk, 0, stream>>>(s2, V2, comp2, bias2, agg2,
                                              /*NB=*/16, /*R=*/16, /*OUT=*/ODIM);
  rgcn_gemm_bf16<false><<<dim3(ODIM / 64, N_NODES / 128), blk, 0, stream>>>(
      h, loop2, agg2, out, ODIM);
}
